// PQConv2d_59708635349054
// MI455X (gfx1250) — matrix operand order repaired and checked by reading, not compiled, not run on hardware
//
#include <hip/hip_runtime.h>

typedef __bf16 bf16_t;
typedef __bf16 v16bf __attribute__((ext_vector_type(16)));
typedef __bf16 v8bf  __attribute__((ext_vector_type(8)));
typedef float  v8f   __attribute__((ext_vector_type(8)));

#define HH 224
#define WW 224
#define CIN 64
#define OUT 64
#define TILE_W 112
#define ROWS 28
#define NTHREADS 224
// LDS geometry (elements)
#define XPX 114                 // pixels per slab: x0-1 .. x0+112
#define XCST 72                 // channel stride (padded; 72*2B keeps 16B alignment, reduces conflicts)
#define SLAB_ELEMS (XPX * XCST) // 8208 bf16
#define WELEMS (9 * 64 * 64)    // 36864 bf16
#define STAG_STRIDE 113         // float stride, odd vs 64 banks

#define WLDS_BYTES (WELEMS * 2)                  // 73728
#define XLDS_BYTES (3 * SLAB_ELEMS * 2)          // 49248
#define STAG_BYTES (OUT * STAG_STRIDE * 4)       // 28928
#define SMEM_BYTES (WLDS_BYTES + XLDS_BYTES + STAG_BYTES)  // 151904

static __device__ __forceinline__ v16bf cat16(v8bf lo, v8bf hi) {
    return __builtin_shufflevector(lo, hi, 0,1,2,3,4,5,6,7,8,9,10,11,12,13,14,15);
}

__global__ __launch_bounds__(NTHREADS, 2)
void pqconv2d_wmma_kernel(const float* __restrict__ x,
                          const int*   __restrict__ idxs,
                          const float* __restrict__ cb,
                          const float* __restrict__ bias,
                          float*       __restrict__ out)
{
    extern __shared__ char smem[];
    bf16_t* wlds = (bf16_t*)smem;                              // [9][64 o][64 c]
    bf16_t* xlds = (bf16_t*)(smem + WLDS_BYTES);               // 3 slabs [px][c]
    float*  stag = (float*)(smem + WLDS_BYTES + XLDS_BYTES);   // [64 o][113]

    const int tid  = threadIdx.x;
    const int lane = tid & 31;
    const int wave = tid >> 5;
    const int x0 = blockIdx.x * TILE_W;     // 0 or 112
    const int y0 = blockIdx.y * ROWS;       // 0..196
    const int n  = blockIdx.z;              // 0..31

    // ---- dequantize weights -> LDS bf16, layout [khw][o][c], o = h*8+d ----
    for (int i = tid; i < WELEMS; i += NTHREADS) {
        int khw = i >> 12;            // /4096
        int o   = (i >> 6) & 63;
        int c   = i & 63;
        int h = o >> 3, d = o & 7;
        int code = idxs[((h * 64 + c) * 3 + (khw / 3)) * 3 + (khw % 3)];
        float v  = cb[(h * 256 + code) * 8 + d];
        wlds[i] = (bf16_t)v;          // i == (khw*64+o)*64+c
    }

    // ---- row loader: global fp32 row -> LDS bf16 slab [px][c] ----
    auto loadrow = [&](int r) {
        int slab = ((r % 3) + 3) % 3;
        bf16_t* xs = xlds + slab * SLAB_ELEMS;
        const bool rin = (r >= 0) && (r < HH);
        for (int i = tid; i < CIN * XPX; i += NTHREADS) {
            int c = i / XPX;
            int p = i - c * XPX;
            int pg = x0 - 1 + p;
            float v = 0.0f;
            if (rin && pg >= 0 && pg < WW)
                v = x[(((size_t)n * CIN + c) * HH + r) * WW + pg];
            xs[p * XCST + c] = (bf16_t)v;
        }
    };

    loadrow(y0 - 1);
    loadrow(y0);

    const int mrow = lane & 15;                  // A matrix row (pixel in wave tile)
    const int kadd = (lane < 16) ? 0 : 8;        // A K-offset per half-wave
    const int o_l  = lane & 15;                  // D: outch lane
    const int pxb  = wave * 16 + ((lane < 16) ? 0 : 8);  // D: pixel base

    for (int y = y0; y < y0 + ROWS; ++y) {
        loadrow(y + 1);

        // prefetch row y+2 while this row's data settles (global_prefetch_b8)
        {
            int r = y + 2;
            if (r < HH) {
                const float* base = x + (((size_t)n * CIN) * HH + r) * WW + x0;
                for (int i = tid; i < CIN * 7; i += NTHREADS) {
                    int c = i / 7, seg = i - (i / 7) * 7;
                    __builtin_prefetch(base + (size_t)c * HH * WW + seg * 16, 0, 0);
                }
            }
        }
        __syncthreads();

        v8f acc[4];
        #pragma unroll
        for (int t = 0; t < 4; ++t) acc[t] = v8f{0.f,0.f,0.f,0.f,0.f,0.f,0.f,0.f};

        #pragma unroll
        for (int kh = 0; kh < 3; ++kh) {
            int slab = (((y - 1 + kh) % 3) + 3) % 3;
            const bf16_t* xs = xlds + slab * SLAB_ELEMS;
            #pragma unroll
            for (int kw = 0; kw < 3; ++kw) {
                const int khw = kh * 3 + kw;
                const int px  = wave * 16 + mrow + kw;   // <= 113
                #pragma unroll
                for (int c0 = 0; c0 < 64; c0 += 32) {
                    // A fragment: x patch, M=pixel, K=channel (pairs per VGPR)
                    const bf16_t* ap = xs + px * XCST + c0 + kadd;
                    v8bf alo = *(const v8bf*)ap;          // K = kadd .. kadd+7
                    v8bf ahi = *(const v8bf*)(ap + 16);   // K = kadd+16 .. kadd+23
                    v16bf A  = cat16(alo, ahi);
                    // B fragments: weights, lane = N (outch), VGPR = K (channel) pairs
                    const bf16_t* wp = wlds + ((size_t)khw * 64 + mrow) * 64 + c0 + kadd;
                    #pragma unroll
                    for (int t = 0; t < 4; ++t) {
                        v8bf blo = *(const v8bf*)(wp + t * 1024);
                        v8bf bhi = *(const v8bf*)(wp + t * 1024 + 16);
                        v16bf B = cat16(blo, bhi);
                        acc[t] = __builtin_amdgcn_wmma_f32_16x16x32_bf16(
                            false, A, false, B, (short)0, acc[t], false, false);
                    }
                }
            }
        }

        // stage D tiles to LDS (lanes hold outch -> transpose for coalesced stores)
        #pragma unroll
        for (int t = 0; t < 4; ++t) {
            #pragma unroll
            for (int j = 0; j < 8; ++j)
                stag[(t * 16 + o_l) * STAG_STRIDE + pxb + j] = acc[t][j];
        }
        __syncthreads();

        // coalesced fp32 store + bias
        for (int i = tid; i < OUT * TILE_W; i += NTHREADS) {
            int o = i / TILE_W;
            int p = i - o * TILE_W;
            out[(((size_t)n * OUT + o) * HH + y) * WW + x0 + p] =
                stag[o * STAG_STRIDE + p] + bias[o];
        }
        __syncthreads();   // protect slab ring + staging before next row
    }
}

extern "C" void kernel_launch(void* const* d_in, const int* in_sizes, int n_in,
                              void* d_out, int out_size, void* d_ws, size_t ws_size,
                              hipStream_t stream) {
    const float* x    = (const float*)d_in[0];
    const int*   idxs = (const int*)d_in[1];
    const float* cb   = (const float*)d_in[2];
    const float* bias = (const float*)d_in[3];
    float* out = (float*)d_out;

    dim3 grid(WW / TILE_W, HH / ROWS, 32);   // (2, 8, 32)
    dim3 block(NTHREADS, 1, 1);              // 7 waves of 32
    pqconv2d_wmma_kernel<<<grid, block, SMEM_BYTES, stream>>>(x, idxs, cb, bias, out);
}
